// VolumeFeatureAggregator_8581344657713
// MI455X (gfx1250) — compile-verified, matrix operand order checked by reading, not run on hardware
//
#include <hip/hip_runtime.h>
#include <hip/hip_bf16.h>

typedef __attribute__((ext_vector_type(16))) _Float16 v16h;
typedef __attribute__((ext_vector_type(8)))  float    v8f;

#define N_PTS   131072
#define C_IN    1018
#define D_IN    1024
#define C_OUT   128
#define GRID_D  32
#define BATCH   16
#define NUM_CELLS (GRID_D*GRID_D*GRID_D)   /* 32768 */
#define NUM_GRIDS (BATCH*NUM_CELLS)        /* 524288 */
#define BN_EPS  1e-5f

// ---- ws layout (float offsets) ----
#define HBUF_OFF   0                       /* h buffer [N,128] fp32            */
#define SUM_OFF    16777216                /* gsum[128] gsumsq[128] scale shift */
#define CNT_OFF    (SUM_OFF + 512)         /* cnt[524288]                      */
#define W1H_OFF    (CNT_OFF + NUM_GRIDS)   /* W1h: 128*1024 f16 col-major      */

__device__ __forceinline__ unsigned lds_addr_of(const void* p) {
  return (unsigned)(unsigned long long)(__attribute__((address_space(3))) const void*)p;
}

// Convert W1 [1024,128] row-major fp32 -> W1h [128][1024] col-major f16
__global__ __launch_bounds__(256) void k_prep_w1(const float* __restrict__ W1,
                                                 _Float16* __restrict__ W1h) {
  int i = blockIdx.x * 256 + threadIdx.x;      // 131072 total
  int c = i & 127, k = i >> 7;
  W1h[c * 1024 + k] = (_Float16)W1[k * 128 + c];
}

// Issue async copies of one 32x128 f16 W-panel into LDS buffer `buf`.
// 512 x 16B chunks, 2 per thread. GVS mode: saddr base + 32-bit vaddr offset.
__device__ __forceinline__ void stage_panel_async(const _Float16* W1h, unsigned wbase,
                                                  int tid, int k0, int buf) {
  unsigned long long gbase = (unsigned long long)W1h;
  #pragma unroll
  for (int it = 0; it < 2; ++it) {
    int i = tid + it * 256;
    int c = i >> 2, part = i & 3;
    unsigned gofs  = (unsigned)(((c << 10) + k0 + part * 8) * 2);
    unsigned laddr = wbase + (unsigned)((buf * 4096 + c * 32 + part * 8) * 2);
    asm volatile("global_load_async_to_lds_b128 %0, %1, %2"
                 :: "v"(laddr), "v"(gofs), "s"(gbase) : "memory");
  }
}

// h = relu(feats @ W1 + b1); accumulate per-channel sum / sumsq for batchnorm
__global__ __launch_bounds__(256) void k_gemm(
    const float* __restrict__ ppf, const float* __restrict__ nocs,
    const float* __restrict__ xyz, const float* __restrict__ b1,
    const _Float16* __restrict__ W1h,
    float* __restrict__ hbuf, float* __restrict__ gsum, float* __restrict__ gsumsq)
{
  __shared__ __align__(16) _Float16 wlds[2][32 * 128]; // double-buffered K-panels
  __shared__ float lsum[C_OUT], lsq[C_OUT];

  const int tid  = threadIdx.x;
  const int wave = tid >> 5;
  const int lane = tid & 31;
  const int lh   = lane & 15;          // A: row within strip; B/D: col within tile
  const int hi   = lane >> 4;          // lane half
  const int koff = hi ? 8 : 0;         // 16-bit A/B layout lane-half K offset
  const int row  = blockIdx.x * 128 + wave * 16 + lh;

  if (tid < C_OUT) { lsum[tid] = 0.f; lsq[tid] = 0.f; }

  const float* rowppf = ppf + (size_t)row * C_IN;
  // tail features in scalar registers (no indexed array -> no scratch)
  float t0, t1, t2, t3, t4, t5;
  {
    float p0 = nocs[row*3+0], p1 = nocs[row*3+1], p2 = nocs[row*3+2];
    float g0 = fminf(fmaxf(rintf(p0 * 31.f), 0.f), 31.f);
    float g1v = fminf(fmaxf(rintf(p1 * 31.f), 0.f), 31.f);
    float g2 = fminf(fmaxf(rintf(p2 * 31.f), 0.f), 31.f);
    t0 = p0 - g0 / 31.f; t1 = p1 - g1v / 31.f; t2 = p2 - g2 / 31.f;
    t3 = xyz[row*3+0]; t4 = xyz[row*3+1]; t5 = xyz[row*3+2];
  }

  v8f acc[8];
  #pragma unroll
  for (int j = 0; j < 8; ++j) { v8f z = {}; acc[j] = z; }

  const unsigned wbase = lds_addr_of(&wlds[0][0]);
  stage_panel_async(W1h, wbase, tid, 0, 0);          // prologue: panel 0 -> buf 0

  for (int kb = 0; kb < 32; ++kb) {
    const int k0 = kb * 32;
    asm volatile("s_wait_asynccnt 0x0" ::: "memory"); // my panel copies landed
    __syncthreads();                                  // everyone's landed
    if (kb < 31) stage_panel_async(W1h, wbase, tid, k0 + 32, (kb + 1) & 1);
    __builtin_prefetch(rowppf + k0 + 32 + koff, 0, 0);

    // ---- build A fragment (ISA 16-bit A 16x32 layout) ----
    union { v16h v; _Float16 e[16]; } a;
    if (k0 != 992) {
      #pragma unroll
      for (int g = 0; g < 2; ++g) {
        int kg = k0 + koff + g * 16;
        #pragma unroll
        for (int i = 0; i < 4; ++i) {
          float2 f = *(const float2*)(rowppf + kg + 2*i);
          a.e[g*8 + 2*i]     = (_Float16)f.x;
          a.e[g*8 + 2*i + 1] = (_Float16)f.y;
        }
      }
    } else {
      // group 0: kg = 992 + koff  (992 / 1000) -> all ppf
      #pragma unroll
      for (int i = 0; i < 4; ++i) {
        float2 f = *(const float2*)(rowppf + 992 + koff + 2*i);
        a.e[2*i] = (_Float16)f.x; a.e[2*i + 1] = (_Float16)f.y;
      }
      // group 1: kg = 1008 + koff -> hi lanes splice offset+xyz at 1016..1023
      if (hi == 0) {
        #pragma unroll
        for (int i = 0; i < 4; ++i) {
          float2 f = *(const float2*)(rowppf + 1008 + 2*i);
          a.e[8 + 2*i] = (_Float16)f.x; a.e[9 + 2*i] = (_Float16)f.y;
        }
      } else {
        float2 f = *(const float2*)(rowppf + 1016);
        a.e[8]  = (_Float16)f.x; a.e[9]  = (_Float16)f.y;
        a.e[10] = (_Float16)t0;  a.e[11] = (_Float16)t1;
        a.e[12] = (_Float16)t2;  a.e[13] = (_Float16)t3;
        a.e[14] = (_Float16)t4;  a.e[15] = (_Float16)t5;
      }
    }

    // ---- 8 col-tiles: B frags from LDS, WMMA ----
    const _Float16* wp0 = &wlds[kb & 1][lh * 32 + koff];
    #pragma unroll
    for (int j = 0; j < 8; ++j) {
      union { v16h v; float4 f4[2]; } b;
      const _Float16* wp = wp0 + j * 16 * 32;
      b.f4[0] = *(const float4*)wp;
      b.f4[1] = *(const float4*)(wp + 16);
      acc[j] = __builtin_amdgcn_wmma_f32_16x16x32_f16(
          false, a.v, false, b.v, (short)0, acc[j], false, false);
    }
  }

  // ---- epilogue: bias + relu, store h, per-channel partial sums ----
  const int rowbase = blockIdx.x * 128 + wave * 16 + hi * 8;  // D layout: M = e + 8*hi
  #pragma unroll
  for (int j = 0; j < 8; ++j) {
    int col = j * 16 + lh;
    float bias = b1[col];
    float s = 0.f, sq = 0.f;
    #pragma unroll
    for (int e = 0; e < 8; ++e) {
      float v = fmaxf(acc[j][e] + bias, 0.f);
      hbuf[(size_t)(rowbase + e) * C_OUT + col] = v;
      s += v; sq += v * v;
    }
    s  += __shfl_xor(s, 16, 32);
    sq += __shfl_xor(sq, 16, 32);
    if (hi == 0) { atomicAdd(&lsum[col], s); atomicAdd(&lsq[col], sq); }
  }
  __syncthreads();
  if (tid < C_OUT) {
    atomicAdd(&gsum[tid],   lsum[tid]);
    atomicAdd(&gsumsq[tid], lsq[tid]);
  }
}

__global__ void k_bnstats(const float* __restrict__ gsum, const float* __restrict__ gsumsq,
                          const float* __restrict__ g1, const float* __restrict__ be1,
                          float* __restrict__ scale, float* __restrict__ shift) {
  int c = threadIdx.x;
  float m   = gsum[c]   * (1.f / N_PTS);
  float var = gsumsq[c] * (1.f / N_PTS) - m * m;
  float inv = rsqrtf(var + BN_EPS);
  float sc  = g1[c] * inv;
  scale[c] = sc;
  shift[c] = be1[c] - m * sc;
}

// one block per point: normalize + scatter-add into transposed volume
__global__ __launch_bounds__(128) void k_scatter(
    const float* __restrict__ hbuf, const float* __restrict__ nocs,
    const int* __restrict__ bidx,
    const float* __restrict__ scale, const float* __restrict__ shift,
    float* __restrict__ out, float* __restrict__ cnt)
{
  const int n = blockIdx.x;
  const int c = threadIdx.x;
  int ix = (int)rintf(nocs[n*3+0] * 31.f); ix = min(max(ix, 0), 31);
  int iy = (int)rintf(nocs[n*3+1] * 31.f); iy = min(max(iy, 0), 31);
  int iz = (int)rintf(nocs[n*3+2] * 31.f); iz = min(max(iz, 0), 31);
  int cell = (ix << 10) | (iy << 5) | iz;
  int b = bidx[n];
  float v = hbuf[(size_t)n * C_OUT + c] * scale[c] + shift[c];
  atomicAdd(&out[(((size_t)b * C_OUT + c) << 15) | (size_t)cell], v);
  if (c == 0) atomicAdd(&cnt[((size_t)b << 15) + cell], 1.0f);
}

__global__ __launch_bounds__(256) void k_div(float* __restrict__ out,
                                             const float* __restrict__ cnt) {
  size_t o = (size_t)blockIdx.x * 256 + threadIdx.x;   // 67,108,864 total
  int b    = (int)(o >> 22);       // 128*32768 = 2^22 per batch
  int cell = (int)(o & 32767);
  float cc = cnt[((size_t)b << 15) + cell];
  out[o] = out[o] / fmaxf(cc, 1.f);
}

extern "C" void kernel_launch(void* const* d_in, const int* in_sizes, int n_in,
                              void* d_out, int out_size, void* d_ws, size_t ws_size,
                              hipStream_t stream) {
  (void)in_sizes; (void)n_in; (void)ws_size;
  const float* xyz  = (const float*)d_in[0];
  const float* nocs = (const float*)d_in[1];
  const float* ppf  = (const float*)d_in[2];
  /* d_in[3] pred_confidence: unused by reference */
  const int*   bidx = (const int*)d_in[4];
  /* d_in[5] batch_size: compile-time 16 */
  const float* W1   = (const float*)d_in[6];
  const float* b1   = (const float*)d_in[7];
  const float* g1   = (const float*)d_in[8];
  const float* be1  = (const float*)d_in[9];

  float* out    = (float*)d_out;
  float* wsf    = (float*)d_ws;
  float* hbuf   = wsf + HBUF_OFF;
  float* gsum   = wsf + SUM_OFF;
  float* gsumsq = gsum + 128;
  float* scale  = gsum + 256;
  float* shift  = gsum + 384;
  float* cnt    = wsf + CNT_OFF;
  _Float16* W1h = (_Float16*)(wsf + W1H_OFF);

  hipMemsetAsync(out,  0, (size_t)out_size * sizeof(float), stream);
  hipMemsetAsync(gsum, 0, (512 + (size_t)NUM_GRIDS) * sizeof(float), stream);

  k_prep_w1<<<512, 256, 0, stream>>>(W1, W1h);
  k_gemm<<<N_PTS / 128, 256, 0, stream>>>(ppf, nocs, xyz, b1, W1h, hbuf, gsum, gsumsq);
  k_bnstats<<<1, 128, 0, stream>>>(gsum, gsumsq, g1, be1, scale, shift);
  k_scatter<<<N_PTS, 128, 0, stream>>>(hbuf, nocs, bidx, scale, shift, out, cnt);
  k_div<<<(BATCH * C_OUT * NUM_CELLS) / 256, 256, 0, stream>>>(out, cnt);
}
